// GraphTransformerLayer_24859270709324
// MI455X (gfx1250) — matrix operand order repaired and checked by reading, not compiled, not run on hardware
//
#include <hip/hip_runtime.h>
#include <hip/hip_bf16.h>
#include <cstddef>

// ---------------------------------------------------------------------------
// GATv2 graph-transformer layer for gfx1250 (wave32, WMMA f16 -> f32 accum)
// A-tile staged in LDS as f16: convert once in the fill, ds_load_b128 fragments
// ---------------------------------------------------------------------------

typedef __attribute__((ext_vector_type(16))) _Float16 v16h;
typedef __attribute__((ext_vector_type(8)))  _Float16 v8h;
typedef __attribute__((ext_vector_type(4)))  _Float16 v4h;
typedef __attribute__((ext_vector_type(8)))  float    v8f;

union F16Frag { v16h v; v8h h[2]; _Float16 e[16]; };

#define LDSH 136   // halves per row: 272 B = 17*16B -> 16B-aligned rows, conflict-free strides

// Y[M x 128] = act(X[M x 128]) @ W[128 x 128] + bias
//   act(x)  = preb ? relu(x + preb) : x          (fused pre-activation, per-column preb[128])
__global__ __launch_bounds__(256)
void gemm128_wmma(const float* __restrict__ X, const float* __restrict__ W,
                  const float* __restrict__ bias, const float* __restrict__ preb,
                  float* __restrict__ Y, int M)
{
    __shared__ _Float16 ldsA[64 * LDSH];
    const int tid   = threadIdx.x;
    const int wave  = tid >> 5;
    const int lane  = tid & 31;
    const int lpos  = lane & 15;
    const int halfw = lane >> 4;
    const int row0  = blockIdx.x * 64;
    const int ncol  = wave * 16 + lpos;
    const bool full = (row0 + 64 <= M);   // uniform: tail-free fast paths

    // ---- preload B fragments (W columns [wave*16, wave*16+16), f32 -> f16) ----
    // B (f16, 32x16) layout: VGPR v holds K = {2v + 8*half (+8 if v>=4), +1}, lane%16 = N
    F16Frag bf[4];
    #pragma unroll
    for (int kk = 0; kk < 4; ++kk) {
        #pragma unroll
        for (int v = 0; v < 8; ++v) {
            const int kb = kk * 32 + 2 * v + 8 * halfw + (v >= 4 ? 8 : 0);
            bf[kk].e[2 * v]     = (_Float16)W[kb * 128 + ncol];
            bf[kk].e[2 * v + 1] = (_Float16)W[(kb + 1) * 128 + ncol];
        }
    }
    const float bv = bias ? bias[ncol] : 0.0f;

    // ---- cooperative load of X tile (64 rows x 128 cols) -> f16 in LDS ----
    if (full) {
        #pragma unroll
        for (int i = tid; i < 64 * 32; i += 256) {
            const int r = i >> 5, kq = i & 31;
            float4 val = ((const float4*)X)[(size_t)(row0 + r) * 32 + kq];
            if (preb) {
                const float4 pb = ((const float4*)preb)[kq];
                val.x = fmaxf(val.x + pb.x, 0.f);
                val.y = fmaxf(val.y + pb.y, 0.f);
                val.z = fmaxf(val.z + pb.z, 0.f);
                val.w = fmaxf(val.w + pb.w, 0.f);
            }
            v4h hv;
            hv[0] = (_Float16)val.x; hv[1] = (_Float16)val.y;
            hv[2] = (_Float16)val.z; hv[3] = (_Float16)val.w;
            *(v4h*)&ldsA[r * LDSH + 4 * kq] = hv;   // 8B-aligned LDS store
        }
    } else {
        #pragma unroll
        for (int i = tid; i < 64 * 32; i += 256) {
            const int r = i >> 5, kq = i & 31;
            const int grow = row0 + r;
            float4 val = make_float4(0.f, 0.f, 0.f, 0.f);
            if (grow < M) {
                val = ((const float4*)X)[(size_t)grow * 32 + kq];
                if (preb) {
                    const float4 pb = ((const float4*)preb)[kq];
                    val.x = fmaxf(val.x + pb.x, 0.f);
                    val.y = fmaxf(val.y + pb.y, 0.f);
                    val.z = fmaxf(val.z + pb.z, 0.f);
                    val.w = fmaxf(val.w + pb.w, 0.f);
                }
            }
            v4h hv;
            hv[0] = (_Float16)val.x; hv[1] = (_Float16)val.y;
            hv[2] = (_Float16)val.z; hv[3] = (_Float16)val.w;
            *(v4h*)&ldsA[r * LDSH + 4 * kq] = hv;
        }
    }
    __syncthreads();

    // ---- 4 row tiles of 16; each wave: 4 WMMAs per tile (K = 128) ----
    #pragma unroll
    for (int t = 0; t < 4; ++t) {
        const _Float16* arow = &ldsA[(t * 16 + lpos) * LDSH];
        v8f acc = {};
        #pragma unroll
        for (int kk = 0; kk < 4; ++kk) {
            // A (f16, 16x32) layout per lane: halves [8h .. 8h+8) then [16+8h .. 16+8h+8)
            // -> exactly two 16B contiguous runs: 2x ds_load_b128, zero conversions
            F16Frag af;
            af.h[0] = *(const v8h*)&arow[kk * 32 + 8 * halfw];
            af.h[1] = *(const v8h*)&arow[kk * 32 + 16 + 8 * halfw];
            acc = __builtin_amdgcn_wmma_f32_16x16x32_f16(false, af.v, false, bf[kk].v,
                                                         (short)0, acc, false, false);
        }
        // C/D layout: VGPR r -> row M = r + 8*half, col N = lane%16
        const int rbase = row0 + t * 16 + 8 * halfw;
        float* yp = Y + (size_t)rbase * 128 + ncol;   // rows r: constant 512B offsets
        if (full) {
            #pragma unroll
            for (int r = 0; r < 8; ++r) yp[(size_t)r * 128] = acc[r] + bv;
        } else {
            #pragma unroll
            for (int r = 0; r < 8; ++r)
                if (rbase + r < M) yp[(size_t)r * 128] = acc[r] + bv;
        }
    }
}

// ---------------------------------------------------------------------------

__global__ void fill_zero_kernel(float* p, size_t n) {
    size_t i = (size_t)blockIdx.x * blockDim.x + threadIdx.x;
    const size_t stride = (size_t)gridDim.x * blockDim.x;
    for (; i < n; i += stride) p[i] = 0.0f;
}

// W_comb[129 x 128]: rows 0..127 = W_ee @ W_edge, row 128 = b_ee @ W_edge
__global__ void combine_w_kernel(const float* __restrict__ W_ee, const float* __restrict__ b_ee,
                                 const float* __restrict__ W_edge,
                                 float* __restrict__ Wc, float* __restrict__ bc) {
    const int gid = blockIdx.x * blockDim.x + threadIdx.x;
    if (gid >= 129 * 128) return;
    const int r = gid >> 7, c = gid & 127;
    const float* xrow = (r < 128) ? (W_ee + r * 128) : b_ee;
    float acc = 0.f;
    for (int k = 0; k < 128; ++k) acc = fmaf(xrow[k], W_edge[k * 128 + c], acc);
    if (r < 128) Wc[r * 128 + c] = acc; else bc[c] = acc;
}

__global__ void deg_kernel(const int* __restrict__ dst, float* __restrict__ deg, int E) {
    const int e = blockIdx.x * blockDim.x + threadIdx.x;
    if (e >= E) return;
    atomicAdd(&deg[dst[e]], 1.0f);
}

// mean[dst] += edge_attr[e]   (thread = (edge, 4-col chunk))
__global__ void meansum_kernel(const float* __restrict__ eattr, const int* __restrict__ dst,
                               float* __restrict__ mean, int E) {
    const int gid = blockIdx.x * blockDim.x + threadIdx.x;
    if (gid >= E * 32) return;
    const int e = gid >> 5, q = gid & 31;
    const int d = dst[e];
    const float4 v = ((const float4*)eattr)[(size_t)e * 32 + q];
    float* o = mean + (size_t)d * 128 + 4 * q;
    atomicAdd(o + 0, v.x); atomicAdd(o + 1, v.y);
    atomicAdd(o + 2, v.z); atomicAdd(o + 3, v.w);
}

__global__ void meandiv_kernel(float* __restrict__ mean, const float* __restrict__ deg, int N) {
    const int gid = blockIdx.x * blockDim.x + threadIdx.x;
    if (gid >= N * 32) return;
    const int n = gid >> 5, q = gid & 31;
    const float dg = fmaxf(deg[n], 1.0f);
    float4 v = ((float4*)mean)[(size_t)n * 32 + q];
    v.x /= dg; v.y /= dg; v.z /= dg; v.w /= dg;
    ((float4*)mean)[(size_t)n * 32 + q] = v;
}

// reference: deg==0 => loop_attr == 0 => its lin_edge projection is 0 (no bias).
// The GEMM wrote b_comb there (meanb row was all-zero); zero those rows out.
__global__ void ee_fixup_kernel(float* __restrict__ eeN, const float* __restrict__ deg, int N) {
    const int gid = blockIdx.x * blockDim.x + threadIdx.x;
    if (gid >= N * 32) return;
    const int n = gid >> 5, q = gid & 31;
    if (deg[n] == 0.0f)
        ((float4*)eeN)[(size_t)n * 32 + q] = make_float4(0.f, 0.f, 0.f, 0.f);
}

// order-preserving encoding for float atomicMax on uint
__device__ inline unsigned fenc(float f) {
    unsigned u = __float_as_uint(f);
    return (u & 0x80000000u) ? ~u : (u | 0x80000000u);
}
__device__ inline float fdec(unsigned u) {
    return __uint_as_float((u & 0x80000000u) ? (u ^ 0x80000000u) : ~u);
}

// one wave per edge-row: logits[e][h] = att_h . leaky_relu(xl[src] + xr[dst] + ee[e])
__global__ __launch_bounds__(256)
void logits_kernel(const float* __restrict__ xl, const float* __restrict__ xr,
                   const float* __restrict__ ee, const int* __restrict__ src,
                   const int* __restrict__ dst, const float* __restrict__ att,
                   float* __restrict__ logits, unsigned* __restrict__ mmax,
                   int E, int T) {
    const int gid  = blockIdx.x * blockDim.x + threadIdx.x;
    const int e    = gid >> 5;          // uniform across the wave
    const int lane = gid & 31;
    if (e >= T) return;
    int sn, dn;
    if (e < E) { sn = src[e]; dn = dst[e]; } else { sn = dn = e - E; }
    const float4 a = ((const float4*)xl)[(size_t)sn * 32 + lane];
    const float4 b = ((const float4*)xr)[(size_t)dn * 32 + lane];
    const float4 c = ((const float4*)ee)[(size_t)e * 32 + lane];
    const float4 w = ((const float4*)att)[lane];
    float z0 = a.x + b.x + c.x; z0 = (z0 >= 0.f) ? z0 : 0.2f * z0;
    float z1 = a.y + b.y + c.y; z1 = (z1 >= 0.f) ? z1 : 0.2f * z1;
    float z2 = a.z + b.z + c.z; z2 = (z2 >= 0.f) ? z2 : 0.2f * z2;
    float z3 = a.w + b.w + c.w; z3 = (z3 >= 0.f) ? z3 : 0.2f * z3;
    float p = z0 * w.x + z1 * w.y + z2 * w.z + z3 * w.w;
    // reduce over the 8 lanes that share one head (lane/8 == head)
    p += __shfl_down(p, 4, 8);
    p += __shfl_down(p, 2, 8);
    p += __shfl_down(p, 1, 8);
    if ((lane & 7) == 0) {
        const int h = lane >> 3;
        logits[(size_t)e * 4 + h] = p;
        atomicMax(&mmax[(size_t)dn * 4 + h], fenc(p));
    }
}

// expv = exp(logit - max[dst]); denom[dst] += expv; logits overwritten in place
__global__ void expdenom_kernel(float* __restrict__ logits, const unsigned* __restrict__ mmax,
                                float* __restrict__ denom, const int* __restrict__ dst,
                                int E, int T) {
    const int gid = blockIdx.x * blockDim.x + threadIdx.x;
    if (gid >= T * 4) return;
    const int e = gid >> 2, h = gid & 3;
    const int dn = (e < E) ? dst[e] : (e - E);
    const float m = fdec(mmax[(size_t)dn * 4 + h]);
    const float ex = expf(logits[gid] - m);
    logits[gid] = ex;
    atomicAdd(&denom[(size_t)dn * 4 + h], ex);
}

// agg[dst] += (expv/denom[dst]) * xl[src]   (thread = (edge-row, 4-col chunk))
__global__ void agg_kernel(const float* __restrict__ xl, const float* __restrict__ expv,
                           const float* __restrict__ denom, const int* __restrict__ src,
                           const int* __restrict__ dst, float* __restrict__ agg,
                           int E, int T) {
    const int gid = blockIdx.x * blockDim.x + threadIdx.x;
    if (gid >= T * 32) return;
    const int e = gid >> 5, q = gid & 31;
    int sn, dn;
    if (e < E) { sn = src[e]; dn = dst[e]; } else { sn = dn = e - E; }
    const int h = q >> 3;
    const float alpha = expv[(size_t)e * 4 + h] / denom[(size_t)dn * 4 + h];
    const float4 v = ((const float4*)xl)[(size_t)sn * 32 + q];
    float* o = agg + (size_t)dn * 128 + 4 * q;
    atomicAdd(o + 0, alpha * v.x); atomicAdd(o + 1, alpha * v.y);
    atomicAdd(o + 2, alpha * v.z); atomicAdd(o + 3, alpha * v.w);
}

// ---------------------------------------------------------------------------

static inline int cdiv(int a, int b) { return (a + b - 1) / b; }

extern "C" void kernel_launch(void* const* d_in, const int* in_sizes, int n_in,
                              void* d_out, int out_size, void* d_ws, size_t ws_size,
                              hipStream_t stream) {
    const float* x      = (const float*)d_in[0];
    const int*   eidx   = (const int*)d_in[1];
    const float* eattr  = (const float*)d_in[2];
    const float* W_ee   = (const float*)d_in[3];
    const float* b_ee   = (const float*)d_in[4];
    const float* W_l    = (const float*)d_in[5];
    const float* b_l    = (const float*)d_in[6];
    const float* W_r    = (const float*)d_in[7];
    const float* b_r    = (const float*)d_in[8];
    const float* W_edge = (const float*)d_in[9];
    const float* att    = (const float*)d_in[10];
    const float* b_conv = (const float*)d_in[11];
    const float* W_lin  = (const float*)d_in[12];
    const float* b_lin  = (const float*)d_in[13];

    const int N = in_sizes[0] / 128;
    const int E = in_sizes[1] / 2;
    const int T = E + N;
    const int* src = eidx;
    const int* dst = eidx + E;

    // ---- workspace carve (floats) ----
    float* w = (float*)d_ws;
    size_t o = 0;
    float*    ee     = w + o; o += (size_t)T * 128;
    float*    xl     = w + o; o += (size_t)N * 128;
    float*    xr     = w + o; o += (size_t)N * 128;
    float*    meanb  = w + o; o += (size_t)N * 128;   // |-- contiguous zero region
    float*    agg    = w + o; o += (size_t)N * 128;   // |
    float*    degp   = w + o; o += (size_t)N;         // |
    unsigned* mmax   = (unsigned*)(w + o); o += (size_t)N * 4; // | (0 == encoded -max)
    float*    denom  = w + o; o += (size_t)N * 4;     // |
    float*    logits = w + o; o += (size_t)T * 4;
    float*    Wcomb  = w + o; o += 128 * 128;
    float*    bcomb  = w + o; o += 128;
    (void)ws_size; (void)n_in; (void)out_size;

    const size_t nzero = (size_t)N * 128 * 2 + (size_t)N + (size_t)N * 4 * 2; // meanb..denom
    fill_zero_kernel<<<4096, 256, 0, stream>>>(meanb, nzero);

    // fused edge weights: Wcomb = W_ee @ W_edge, bcomb = b_ee @ W_edge
    combine_w_kernel<<<cdiv(129 * 128, 256), 256, 0, stream>>>(W_ee, b_ee, W_edge, Wcomb, bcomb);

    // degree + per-dst mean of edge_attr (linearity lets us defer the linear layers)
    deg_kernel<<<cdiv(E, 256), 256, 0, stream>>>(dst, degp, E);
    meansum_kernel<<<cdiv(E * 32, 256), 256, 0, stream>>>(eattr, dst, meanb, E);
    meandiv_kernel<<<cdiv(N * 32, 256), 256, 0, stream>>>(meanb, degp, N);

    // node projections
    gemm128_wmma<<<cdiv(N, 64), 256, 0, stream>>>(x, W_l, b_l, nullptr, xl, N);
    gemm128_wmma<<<cdiv(N, 64), 256, 0, stream>>>(x, W_r, b_r, nullptr, xr, N);

    // edge projections: ee[0:E] from edge_attr, ee[E:T] from per-dst mean
    gemm128_wmma<<<cdiv(E, 64), 256, 0, stream>>>(eattr, Wcomb, bcomb, nullptr, ee, E);
    gemm128_wmma<<<cdiv(N, 64), 256, 0, stream>>>(meanb, Wcomb, bcomb, nullptr,
                                                  ee + (size_t)E * 128, N);
    // deg==0 rows: reference projects a zero loop_attr -> zero (no bias)
    ee_fixup_kernel<<<cdiv(N * 32, 256), 256, 0, stream>>>(ee + (size_t)E * 128, degp, N);

    // attention logits + segment max
    logits_kernel<<<cdiv(T * 32, 256), 256, 0, stream>>>(xl, xr, ee, src, dst, att,
                                                         logits, mmax, E, T);
    // segment softmax denominator (logits -> expv in place)
    expdenom_kernel<<<cdiv(T * 4, 256), 256, 0, stream>>>(logits, mmax, denom, dst, E, T);
    // weighted scatter-add aggregation
    agg_kernel<<<cdiv(T * 32, 256), 256, 0, stream>>>(xl, logits, denom, src, dst, agg, E, T);

    // out = relu(agg + b_conv) @ W_lin + b_lin
    gemm128_wmma<<<cdiv(N, 64), 256, 0, stream>>>(agg, W_lin, b_lin, b_conv,
                                                  (float*)d_out, N);
}
